// AttentionOperator_43069932044621
// MI455X (gfx1250) — matrix operand order repaired and checked by reading, not compiled, not run on hardware
//
#include <hip/hip_runtime.h>

typedef __attribute__((ext_vector_type(16))) _Float16 v16h;
typedef __attribute__((ext_vector_type(8)))  _Float16 v8h;
typedef __attribute__((ext_vector_type(8)))  float    v8f;
typedef __attribute__((ext_vector_type(4)))  float    v4f;

#define N_TOK 4096
#define BATCH 4
#define LOG2E 1.4426950408889634f

// ---- pre-pass: per-batch max/min of x, plus the two 16-dim dot products ----
__global__ void attn_prep_kernel(const float* __restrict__ x,
                                 const float* __restrict__ wq,
                                 const float* __restrict__ wk,
                                 const float* __restrict__ wv,
                                 const float* __restrict__ wo,
                                 float* __restrict__ ws) {
    __shared__ float smax[256];
    __shared__ float smin[256];
    const int t  = threadIdx.x;
    const int b  = t >> 6;     // 4 batches x 64 threads
    const int i0 = t & 63;
    float mx = -3.4e38f, mn = 3.4e38f;
    for (int i = i0; i < N_TOK; i += 64) {
        float v = x[b * N_TOK + i];
        mx = fmaxf(mx, v);
        mn = fminf(mn, v);
    }
    smax[t] = mx;
    smin[t] = mn;
    __syncthreads();
    for (int s = 32; s > 0; s >>= 1) {
        if (i0 < s) {
            smax[t] = fmaxf(smax[t], smax[t + s]);
            smin[t] = fminf(smin[t], smin[t + s]);
        }
        __syncthreads();
    }
    if (i0 == 0) {
        ws[2 + b] = smax[t];
        ws[6 + b] = smin[t];
    }
    if (t == 0) {
        float dqk = 0.f, dvo = 0.f;
        for (int i = 0; i < 16; ++i) {
            dqk += wq[i] * wk[i];
            dvo += wv[i] * wo[i];
        }
        ws[0] = 0.25f * dqk;   // scale = 1/sqrt(16), TAU = 1
        ws[1] = dvo;           // w_v . w_out
    }
}

// ---- main kernel: 4 waves/block, each wave owns 16 rows n, loops m in 32-tiles ----
__global__ void __launch_bounds__(128)
attn_main_kernel(const float* __restrict__ x,
                 const float* __restrict__ ws,
                 float* __restrict__ out) {
    __shared__ __align__(16) float    xs[N_TOK];  // f32 copy (exp inputs)
    __shared__ __align__(16) _Float16 xh[N_TOK];  // f16 copy (A-matrix values)

    const int b = blockIdx.y;
    const int t = threadIdx.x;

    for (int i = t; i < N_TOK; i += 128) {
        float v = x[b * N_TOK + i];
        xs[i] = v;
        xh[i] = (_Float16)v;
    }
    __syncthreads();

    const float cc  = ws[0];
    const float dvo = ws[1];
    const float xmx = ws[2 + b];
    const float xmn = ws[6 + b];

    const int wave = t >> 5;
    const int lane = t & 31;
    const int row  = lane & 15;   // WMMA M (A) / N (B,D) index
    const int kh   = lane >> 4;   // which K-half this lane covers
    const int n    = blockIdx.x * 64 + wave * 16 + row;

    // a = cc * x_n ; row max of scores = max(a*xmax, a*xmin); work in exp2 domain
    const float a  = cc * xs[n];
    const float rm = fmaxf(a * xmx, a * xmn);
    const float aL = a * LOG2E;
    const float rL = rm * LOG2E;

    const _Float16 oneh = (_Float16)1.0f;
    const _Float16 zh   = (_Float16)0.0f;

    v8f acc = {};  // D rows: M=0 -> sum(x*e), M=1 -> sum(e); rest stay 0

    for (int mt = 0; mt < N_TOK; mt += 32) {
        // ---- B tile (32x16 f16): lane holds column N=row, K = h + 16*kh ----
        // => e_{n, mt + 16*kh + h} for h = 0..15 : 16 contiguous x reads + 16 exps
        const v4f* px = (const v4f*)(xs + mt + 16 * kh);
        v16h Bm;
#pragma unroll
        for (int q = 0; q < 4; ++q) {
            v4f xv = px[q];
#pragma unroll
            for (int j = 0; j < 4; ++j) {
                float e = __builtin_amdgcn_exp2f(fmaf(aL, xv[j], -rL));
                Bm[q * 4 + j] = (_Float16)e;
            }
        }

        // ---- A tile (16x32 f16): row 0 = x values, row 1 = ones, rest 0 ----
        // lane's K set: K = h + 16*kh for h = 0..15
        const v8h lo = *(const v8h*)(xh + mt + 16 * kh);
        const v8h hi = *(const v8h*)(xh + mt + 16 * kh + 8);
        v16h Am;
#pragma unroll
        for (int h = 0; h < 8; ++h) {
            Am[h]     = (row == 0) ? lo[h] : ((row == 1) ? oneh : zh);
            Am[h + 8] = (row == 0) ? hi[h] : ((row == 1) ? oneh : zh);
        }

        // D += A x B : accumulates numerator (M=0) and denominator (M=1) per column n
        acc = __builtin_amdgcn_wmma_f32_16x16x32_f16(
            false, Am, false, Bm, (short)0, acc, false, false);
    }

    // lanes 0..15 hold D rows M=0 (elem 0) and M=1 (elem 1) for column N=lane
    if (kh == 0) {
        float num = acc[0];
        float den = acc[1];
        out[b * N_TOK + n] = dvo * num / den;
    }
}

extern "C" void kernel_launch(void* const* d_in, const int* in_sizes, int n_in,
                              void* d_out, int out_size, void* d_ws, size_t ws_size,
                              hipStream_t stream) {
    const float* x  = (const float*)d_in[0];
    const float* wq = (const float*)d_in[1];
    const float* wk = (const float*)d_in[2];
    const float* wv = (const float*)d_in[3];
    const float* wo = (const float*)d_in[4];
    float* ws  = (float*)d_ws;
    float* o   = (float*)d_out;

    attn_prep_kernel<<<1, 256, 0, stream>>>(x, wq, wk, wv, wo, ws);
    attn_main_kernel<<<dim3(N_TOK / 64, BATCH), 128, 0, stream>>>(x, ws, o);
}
